// MultiheadAttention_63118839382477
// MI455X (gfx1250) — matrix operand order repaired and checked by reading, not compiled, not run on hardware
//
#include <hip/hip_runtime.h>
#include <hip/hip_bf16.h>
#include <stdint.h>
#include <math.h>

// ---------------------------------------------------------------------------
// MI455X (gfx1250) multi-head attention forward, bf16 WMMA pipeline:
//   1) fp32 -> bf16 convert (x, w_qkv, w_out)
//   2) QKV = x @ w_qkv^T + b  (bf16 WMMA, async-LDS double-buffered GEMM)
//   3) flash attention (causal, online softmax, WMMA for QK^T and PV,
//      async global->LDS K-tile fetch)
//   4) out = attn @ w_out^T + b (bf16 WMMA, fp32 output)
// ---------------------------------------------------------------------------

typedef __attribute__((ext_vector_type(16))) __bf16 v16bf;
typedef __attribute__((ext_vector_type(8)))  __bf16 v8bf;
typedef __attribute__((ext_vector_type(4)))  __bf16 v4bf;
typedef __attribute__((ext_vector_type(8)))  float  v8f;

union Frag16 {
  v16bf v;
  v8bf  h[2];
};

__device__ __forceinline__ v8f wmma_bf16(v16bf a, v16bf b, v8f c) {
  // (neg_a, A, neg_b, B, c_mod, C, reuse_a, reuse_b)
  return __builtin_amdgcn_wmma_f32_16x16x32_bf16(false, a, false, b, (short)0, c,
                                                 false, false);
}

// ---- async global -> LDS copy (CDNA5 GLOBAL_LOAD_ASYNC_TO_LDS_B128) ----
// The clang builtin wants LangAS-qualified (__device__/__shared__) int4
// pointers that HIP source cannot spell, so issue the instruction via inline
// asm (GV mode: vdst = per-lane LDS byte offset, vaddr = 64-bit global addr).
#define USE_ASYNC_LDS 1

#define LAS __attribute__((address_space(3)))

// copy 16 bytes (8 bf16) from global to LDS
__device__ __forceinline__ void copy16(const __bf16* g, __bf16* l) {
#if USE_ASYNC_LDS
  // generic->local addrspacecast == truncate to the 32-bit LDS byte offset
  uint32_t loff = (uint32_t)(size_t)(LAS __bf16*)l;
  asm volatile("global_load_async_to_lds_b128 %0, %1, off"
               :: "v"(loff), "v"((uint64_t)(size_t)g)
               : "memory");
#else
  *(v8bf*)l = *(const v8bf*)g;
#endif
}

__device__ __forceinline__ void wait_async0() {
#if USE_ASYNC_LDS
  asm volatile("s_wait_asynccnt 0x0" ::: "memory");
#endif
}

constexpr int Bsz = 2, Ssz = 2048, Dsz = 1024, Hh = 16, HD = 64;
constexpr int Mrows = Bsz * Ssz;  // 4096

// ---------------- fp32 -> bf16 conversion ----------------
__global__ void cvt_bf16_kernel(const float* __restrict__ src,
                                __bf16* __restrict__ dst, int n) {
  int i = (blockIdx.x * blockDim.x + threadIdx.x) * 4;
  if (i >= n) return;
  float4 f = *(const float4*)(src + i);
  v4bf o;
  o[0] = (__bf16)f.x; o[1] = (__bf16)f.y; o[2] = (__bf16)f.z; o[3] = (__bf16)f.w;
  *(v4bf*)(dst + i) = o;
}

// ---------------- NT GEMM: C[M,N] = A[M,K] @ Bw[N,K]^T + bias ----------------
// 128x128 tile per workgroup, 8 waves (4 in M x 2 in N), each wave 32x64.
// Double-buffered LDS; tile i+1 fetched with async global->LDS copies while
// tile i is consumed by v_wmma_f32_16x16x32_bf16; one barrier per K-step.
template <bool OUT_BF16>
__global__ __launch_bounds__(256) void gemm_nt_kernel(
    const __bf16* __restrict__ A, const __bf16* __restrict__ Bw,
    const float* __restrict__ bias, void* __restrict__ Cout, int M, int N,
    int K) {
  constexpr int KB = 32;
  constexpr int LST = KB + 8;  // pad: 80B row stride, keeps 16B alignment
  __shared__ __bf16 As[2][128][LST];
  __shared__ __bf16 Bs[2][128][LST];

  const int tid  = threadIdx.x;
  const int lane = tid & 31;
  const int wav  = tid >> 5;   // 0..7
  const int wm   = wav & 3;    // 4 waves along M
  const int wn   = wav >> 2;   // 2 waves along N
  const int hlf  = lane >> 4;
  const int lnl  = lane & 15;
  const int mblk = blockIdx.y * 128;
  const int nblk = blockIdx.x * 128;

  // fetch one 128x32 A tile + 128x32 B tile: 512 16B chunks, 2 per thread each
  auto issue = [&](int kt, int buf) {
#pragma unroll
    for (int i = 0; i < 2; ++i) {
      int c = tid + i * 256;            // 0..511
      int r = c >> 2, kc = (c & 3) * 8; // 8 bf16 = 16B
      copy16(&A[(size_t)(mblk + r) * K + kt + kc], &As[buf][r][kc]);
      copy16(&Bw[(size_t)(nblk + r) * K + kt + kc], &Bs[buf][r][kc]);
    }
  };

  v8f acc[2][4] = {};

  const int NT = K / KB;
  issue(0, 0);
  for (int it = 0; it < NT; ++it) {
    wait_async0();        // this wave's tile-it copies have landed in LDS
    __syncthreads();      // everyone's have; prev compute done reading buf^1
    if (it + 1 < NT) issue((it + 1) * KB, (it + 1) & 1);
    const int buf = it & 1;

    Frag16 fa[2], fb[4];
#pragma unroll
    for (int mi = 0; mi < 2; ++mi) {
      // A 16x32 bf16 layout: lane<16 -> K 0..15; lane>=16 -> 16..31
      const __bf16* p = &As[buf][wm * 32 + mi * 16 + lnl][hlf * 16];
      fa[mi].h[0] = *(const v8bf*)p;
      fa[mi].h[1] = *(const v8bf*)(p + 8);
    }
#pragma unroll
    for (int nj = 0; nj < 4; ++nj)  // B 32x16: lane<16 -> K 0..15; lane>=16 -> 16..31
      fb[nj].v = *(const v16bf*)&Bs[buf][wn * 64 + nj * 16 + lnl][hlf * 16];

#pragma unroll
    for (int mi = 0; mi < 2; ++mi)
#pragma unroll
      for (int nj = 0; nj < 4; ++nj)
        acc[mi][nj] = wmma_bf16(fa[mi].v, fb[nj].v, acc[mi][nj]);
  }

  // epilogue: C layout lane = N, VGPR r -> M = r + 8*half
#pragma unroll
  for (int mi = 0; mi < 2; ++mi)
#pragma unroll
    for (int nj = 0; nj < 4; ++nj)
#pragma unroll
      for (int r = 0; r < 8; ++r) {
        int gr = mblk + wm * 32 + mi * 16 + 8 * hlf + r;
        int gc = nblk + wn * 64 + nj * 16 + lnl;
        float v = acc[mi][nj][r] + bias[gc];
        if constexpr (OUT_BF16)
          ((__bf16*)Cout)[(size_t)gr * N + gc] = (__bf16)v;
        else
          ((float*)Cout)[(size_t)gr * N + gc] = v;
      }
}

// ---------------- flash attention ----------------
// grid = (S/128, B*H); block = 128 threads = 4 waves; wave owns 32 q rows.
// qkv: [B*S][3*D], col = which*D + h*HD + hd. out: [B*S][D] bf16 (= [b,s,h,hd]).
__global__ __launch_bounds__(128) void flash_attn_kernel(
    const __bf16* __restrict__ qkv, __bf16* __restrict__ outb) {
  const int bh = blockIdx.y;
  const int b = bh >> 4, h = bh & 15;
  const int q0 = blockIdx.x * 128;
  const int tid = threadIdx.x, lane = tid & 31, wav = tid >> 5;
  const int hlf = lane >> 4, lnl = lane & 15;
  const int qb = q0 + wav * 32;

  __shared__ __bf16 Kt[32][64];      // K tile, row = kv, K-contiguous
  __shared__ __bf16 Vt[64][40];      // V tile transposed: Vt[hd][kv]
  __shared__ __bf16 Pl[4][32][40];   // per-wave P staging (C-layout -> A-layout)

  const size_t rowstride = 3 * Dsz;
  const size_t base = (size_t)b * Ssz * rowstride;

  // Q fragments held in registers for the whole KV loop (2 M-tiles x 2 K-depth)
  Frag16 fq[2][2];
#pragma unroll
  for (int mi = 0; mi < 2; ++mi)
#pragma unroll
    for (int kf = 0; kf < 2; ++kf) {
      const __bf16* p = qkv + base + (size_t)(qb + mi * 16 + lnl) * rowstride +
                        h * HD + kf * 32 + hlf * 16;
      fq[mi][kf].h[0] = *(const v8bf*)p;
      fq[mi][kf].h[1] = *(const v8bf*)(p + 8);
    }

  v8f o[2][4] = {};
  float mrow[2][8], lrow[2][8];
#pragma unroll
  for (int mi = 0; mi < 2; ++mi)
#pragma unroll
    for (int r = 0; r < 8; ++r) { mrow[mi][r] = -INFINITY; lrow[mi][r] = 0.f; }

  const int ntiles = q0 / 32 + 4;   // workgroup causal bound
  const int wtiles = qb / 32 + 1;   // this wave's causal bound
  for (int j = 0; j < ntiles; ++j) {
    const int kv0 = j * 32;
    {  // cooperative K (async) / V (transposed on store) tile load
      int r = tid >> 2, c = (tid & 3) * 16;
      const __bf16* pk = qkv + base + (size_t)(kv0 + r) * rowstride + Dsz + h * HD + c;
      const __bf16* pv = qkv + base + (size_t)(kv0 + r) * rowstride + 2 * Dsz + h * HD + c;
      copy16(pk, &Kt[r][c]);
      copy16(pk + 8, &Kt[r][c + 8]);
      v8bf va = *(const v8bf*)pv, vb = *(const v8bf*)(pv + 8);
#pragma unroll
      for (int i = 0; i < 8; ++i) {
        Vt[c + i][r] = va[i];
        Vt[c + 8 + i][r] = vb[i];
      }
      wait_async0();
    }
    __syncthreads();

    if (j < wtiles) {
      Frag16 fk[2][2];
#pragma unroll
      for (int nj = 0; nj < 2; ++nj)
#pragma unroll
        for (int kf = 0; kf < 2; ++kf)
          fk[nj][kf].v = *(const v16bf*)&Kt[nj * 16 + lnl][kf * 32 + hlf * 16];

      v8f s[2][2] = {};
#pragma unroll
      for (int mi = 0; mi < 2; ++mi)
#pragma unroll
        for (int nj = 0; nj < 2; ++nj)
#pragma unroll
          for (int kf = 0; kf < 2; ++kf)
            s[mi][nj] = wmma_bf16(fq[mi][kf].v, fk[nj][kf].v, s[mi][nj]);

      // scale by 1/sqrt(HD) + causal mask
#pragma unroll
      for (int mi = 0; mi < 2; ++mi)
#pragma unroll
        for (int nj = 0; nj < 2; ++nj)
#pragma unroll
          for (int r = 0; r < 8; ++r) {
            int row = qb + mi * 16 + 8 * hlf + r;
            int col = kv0 + nj * 16 + lnl;
            float sv = s[mi][nj][r] * 0.125f;
            s[mi][nj][r] = (col <= row) ? sv : -3.0e38f;
          }

      // online softmax: row max via shuffles within 16-lane halves
      float mcur[2][8], scalef[2][8];
#pragma unroll
      for (int mi = 0; mi < 2; ++mi)
#pragma unroll
        for (int r = 0; r < 8; ++r) {
          float v = fmaxf(s[mi][0][r], s[mi][1][r]);
          v = fmaxf(v, __shfl_xor(v, 1, 32));
          v = fmaxf(v, __shfl_xor(v, 2, 32));
          v = fmaxf(v, __shfl_xor(v, 4, 32));
          v = fmaxf(v, __shfl_xor(v, 8, 32));
          float mn = fmaxf(mrow[mi][r], v);
          scalef[mi][r] = __expf(mrow[mi][r] - mn);
          mcur[mi][r] = mn;
          mrow[mi][r] = mn;
        }

#pragma unroll
      for (int mi = 0; mi < 2; ++mi)
#pragma unroll
        for (int r = 0; r < 8; ++r) {
          lrow[mi][r] *= scalef[mi][r];
#pragma unroll
          for (int n = 0; n < 4; ++n) o[mi][n][r] *= scalef[mi][r];
        }

      float lsum[2][8];
#pragma unroll
      for (int mi = 0; mi < 2; ++mi)
#pragma unroll
        for (int r = 0; r < 8; ++r) lsum[mi][r] = 0.f;

#pragma unroll
      for (int mi = 0; mi < 2; ++mi)
#pragma unroll
        for (int nj = 0; nj < 2; ++nj)
#pragma unroll
          for (int r = 0; r < 8; ++r) {
            float p = __expf(s[mi][nj][r] - mcur[mi][r]);
            lsum[mi][r] += p;
            Pl[wav][mi * 16 + 8 * hlf + r][nj * 16 + lnl] = (__bf16)p;
          }

#pragma unroll
      for (int mi = 0; mi < 2; ++mi)
#pragma unroll
        for (int r = 0; r < 8; ++r) {
          float v = lsum[mi][r];
          v += __shfl_xor(v, 1, 32);
          v += __shfl_xor(v, 2, 32);
          v += __shfl_xor(v, 4, 32);
          v += __shfl_xor(v, 8, 32);
          lrow[mi][r] += v;
        }

      // same-wave LDS RAW: LDS is in-order per wave; explicit dscnt drain
      asm volatile("s_wait_dscnt 0x0" ::: "memory");

      Frag16 fp[2], fv[4];
#pragma unroll
      for (int mi = 0; mi < 2; ++mi) {
        const __bf16* p = &Pl[wav][mi * 16 + lnl][hlf * 16];
        fp[mi].h[0] = *(const v8bf*)p;
        fp[mi].h[1] = *(const v8bf*)(p + 8);
      }
#pragma unroll
      for (int n = 0; n < 4; ++n)
        fv[n].v = *(const v16bf*)&Vt[n * 16 + lnl][hlf * 16];

#pragma unroll
      for (int mi = 0; mi < 2; ++mi)
#pragma unroll
        for (int n = 0; n < 4; ++n)
          o[mi][n] = wmma_bf16(fp[mi].v, fv[n].v, o[mi][n]);
    }
    __syncthreads();
  }

  // normalize and store bf16 attention output as [b, s, h, hd] = [4096][1024]
#pragma unroll
  for (int mi = 0; mi < 2; ++mi)
#pragma unroll
    for (int n = 0; n < 4; ++n)
#pragma unroll
      for (int r = 0; r < 8; ++r) {
        int row = qb + mi * 16 + 8 * hlf + r;
        float v = o[mi][n][r] / lrow[mi][r];
        outb[(size_t)(b * Ssz + row) * Dsz + h * HD + n * 16 + lnl] = (__bf16)v;
      }
}

// ---------------- launcher ----------------
extern "C" void kernel_launch(void* const* d_in, const int* in_sizes, int n_in,
                              void* d_out, int out_size, void* d_ws,
                              size_t ws_size, hipStream_t stream) {
  (void)in_sizes; (void)n_in; (void)out_size; (void)ws_size;
  const float* x     = (const float*)d_in[0];
  // d_in[1] = causal mask (int) — causality is hardcoded in the kernel
  const float* w_qkv = (const float*)d_in[2];
  const float* b_qkv = (const float*)d_in[3];
  const float* w_out = (const float*)d_in[4];
  const float* b_out = (const float*)d_in[5];

  auto align = [](size_t v) { return (v + 255) & ~(size_t)255; };
  char* p = (char*)d_ws;
  __bf16* xb    = (__bf16*)p; p += align((size_t)Mrows * Dsz * 2);        // 8 MB
  __bf16* wqkvb = (__bf16*)p; p += align((size_t)3 * Dsz * Dsz * 2);     // 6 MB
  __bf16* woutb = (__bf16*)p; p += align((size_t)Dsz * Dsz * 2);         // 2 MB
  __bf16* qkvb  = (__bf16*)p; p += align((size_t)Mrows * 3 * Dsz * 2);   // 24 MB
  __bf16* attnb = (__bf16*)p; p += align((size_t)Mrows * Dsz * 2);       // 8 MB

  auto cvt = [&](const float* s, __bf16* d, int n) {
    cvt_bf16_kernel<<<dim3((n / 4 + 255) / 256), dim3(256), 0, stream>>>(s, d, n);
  };
  cvt(x, xb, Mrows * Dsz);
  cvt(w_qkv, wqkvb, 3 * Dsz * Dsz);
  cvt(w_out, woutb, Dsz * Dsz);

  // QKV projection: [4096,3072] = xb @ wqkvb^T + b_qkv (bf16 out)
  gemm_nt_kernel<true><<<dim3(3 * Dsz / 128, Mrows / 128), dim3(256), 0, stream>>>(
      xb, wqkvb, b_qkv, qkvb, Mrows, 3 * Dsz, Dsz);

  // causal flash attention over 32 (b,h) pairs
  flash_attn_kernel<<<dim3(Ssz / 128, Bsz * Hh), dim3(128), 0, stream>>>(qkvb, attnb);

  // output projection: fp32 result to d_out
  gemm_nt_kernel<false><<<dim3(Dsz / 128, Mrows / 128), dim3(256), 0, stream>>>(
      attnb, woutb, b_out, d_out, Mrows, Dsz, Dsz);
}
